// Linear_14912126452257
// MI455X (gfx1250) — matrix operand order repaired and checked by reading, not compiled, not run on hardware
//
#include <hip/hip_runtime.h>
#include <hip/hip_bf16.h>

// ---------------------------------------------------------------------------
// SHIRA-linear: out = x @ (W + scatter(coo))^T + bias
//   x: [4,2048,4096] f32   W: [4096,4096] f32   bias: [4096] f32
// GEMM: M=8192, N=4096, K=4096 -> split-f16 WMMA (hi*hi + hi*lo + lo*hi), fp32 acc
// Double-buffered LDS (1 barrier/iter), register prefetch of next tile,
// global_prefetch hints for tile k+2. W' (64MB) is L2-resident on MI455X (192MB).
// ---------------------------------------------------------------------------

typedef __attribute__((ext_vector_type(16))) _Float16 v16h;
typedef __attribute__((ext_vector_type(8)))  float    v8f;

#define IN_F   4096
#define OUT_F  4096
#define BM     128
#define BN     128
#define BK     32
#define NK     (IN_F / BK)      // 128
#define LDSS   40               // LDS row stride in halves (32 + 8 pad, 16B-aligned rows)
#define TILEH  (BM * LDSS)      // 5120 halves per (sub)tile array
#define BUFH   (4 * TILEH)      // sAh,sAl,sBh,sBl per buffer
#define LDS_BYTES (2 * BUFH * 2)  // 81920 B double-buffered

// ---- kernel 1: copy fp32 weight into workspace ----------------------------
__global__ __launch_bounds__(256)
void wcopy_kernel(float* __restrict__ dst, const float* __restrict__ src) {
    size_t i = (size_t)blockIdx.x * 256 + threadIdx.x;
    ((float4*)dst)[i] = ((const float4*)src)[i];
}

// ---- kernel 2: COO scatter-add (SCALING = 1.0) ----------------------------
__global__ __launch_bounds__(256)
void scatter_kernel(float* __restrict__ w, const float* __restrict__ sw,
                    const int* __restrict__ idx, int nnz) {
    int i = blockIdx.x * 256 + threadIdx.x;
    if (i < nnz) {
        int r = idx[i];
        int c = idx[nnz + i];
        atomicAdd(&w[(size_t)r * IN_F + c], sw[i]);
    }
}

// ---- helpers --------------------------------------------------------------
union Pack4 { _Float16 h[4]; uint2 u; };

__device__ __forceinline__ void load_tiles(float4 (&ra)[4], float4 (&rb)[4],
                                           const float* __restrict__ x,
                                           const float* __restrict__ w,
                                           int mBase, int nBase, int kBase, int t) {
#pragma unroll
    for (int i = 0; i < 4; ++i) {
        int idx = t + i * 256;
        int row = idx >> 3;
        int kq  = idx & 7;
        ra[i] = *(const float4*)(x + (size_t)(mBase + row) * IN_F + kBase + kq * 4);
        rb[i] = *(const float4*)(w + (size_t)(nBase + row) * IN_F + kBase + kq * 4);
    }
}

__device__ __forceinline__ void split4(const float4& v, Pack4& H, Pack4& L) {
    _Float16 h;
    h = (_Float16)v.x; H.h[0] = h; L.h[0] = (_Float16)(v.x - (float)h);
    h = (_Float16)v.y; H.h[1] = h; L.h[1] = (_Float16)(v.y - (float)h);
    h = (_Float16)v.z; H.h[2] = h; L.h[2] = (_Float16)(v.z - (float)h);
    h = (_Float16)v.w; H.h[3] = h; L.h[3] = (_Float16)(v.w - (float)h);
}

__device__ __forceinline__ void stage_tiles(_Float16* __restrict__ sAh,
                                            _Float16* __restrict__ sAl,
                                            _Float16* __restrict__ sBh,
                                            _Float16* __restrict__ sBl,
                                            const float4 (&ra)[4],
                                            const float4 (&rb)[4], int t) {
#pragma unroll
    for (int i = 0; i < 4; ++i) {
        int idx = t + i * 256;
        int row = idx >> 3;
        int kq  = idx & 7;
        int off = row * LDSS + kq * 4;
        Pack4 H, L;
        split4(ra[i], H, L);
        *(uint2*)&sAh[off] = H.u;
        *(uint2*)&sAl[off] = L.u;
        split4(rb[i], H, L);
        *(uint2*)&sBh[off] = H.u;
        *(uint2*)&sBl[off] = L.u;
    }
}

// ---- kernel 3: split-f16 WMMA GEMM, double-buffered LDS -------------------
__global__ __launch_bounds__(256)
void gemm_kernel(const float* __restrict__ x,    // [M][K]
                 const float* __restrict__ w,    // [N][K] (post-scatter)
                 const float* __restrict__ bias, // [N]
                 float* __restrict__ out) {      // [M][N]
    extern __shared__ __align__(16) _Float16 smem[];

    const int t      = threadIdx.x;
    const int lane   = t & 31;
    const int wid    = t >> 5;
    const int wm     = wid & 1;         // 2 wave rows  (64 M each)
    const int wn     = wid >> 1;        // 4 wave cols  (32 N each)
    const int lm     = lane & 15;
    const int hiHalf = lane >> 4;

    const int nBase = blockIdx.x * BN;
    const int mBase = blockIdx.y * BM;

    v8f acc[4][2];
    const v8f vzero = {0.f, 0.f, 0.f, 0.f, 0.f, 0.f, 0.f, 0.f};
#pragma unroll
    for (int mt = 0; mt < 4; ++mt)
#pragma unroll
        for (int nt = 0; nt < 2; ++nt) acc[mt][nt] = vzero;

    // prologue: tile 0 -> buffer 0
    float4 ra[4], rb[4];
    load_tiles(ra, rb, x, w, mBase, nBase, 0, t);
    stage_tiles(smem, smem + TILEH, smem + 2 * TILEH, smem + 3 * TILEH, ra, rb, t);
    __syncthreads();

    const int offA = hiHalf * 16;
    const int offB = hiHalf * 16;

    for (int kb = 0; kb < NK; ++kb) {
        const int cur = kb & 1;
        const bool has_next = (kb + 1) < NK;

        // register prefetch of next tile (global latency hidden behind WMMAs)
        if (has_next) load_tiles(ra, rb, x, w, mBase, nBase, (kb + 1) * BK, t);
        // L2 warm-up hint for tile kb+2  -> global_prefetch_b8
        if (kb + 2 < NK) {
            int pk = (kb + 2) * BK + (t & 7) * 4;
            __builtin_prefetch(x + (size_t)(mBase + (t >> 3)) * IN_F + pk, 0, 1);
            __builtin_prefetch(w + (size_t)(nBase + (t >> 3)) * IN_F + pk, 0, 1);
        }

        // ---- compute from buffer `cur` ----
        _Float16* sAh = smem + cur * BUFH;
        _Float16* sAl = sAh + TILEH;
        _Float16* sBh = sAl + TILEH;
        _Float16* sBl = sBh + TILEH;

        // A lane layout: halves[0..15]=K off..off+15                     (off=hiHalf*16)
        // B lane layout: halves[0..15]=K off..off+15                     (off=hiHalf*16)
        union FragU { v16h v; uint4 q[2]; };
        FragU ah[4], al[4], bh[2], bl[2];
#pragma unroll
        for (int mt = 0; mt < 4; ++mt) {
            int mrow = (wm * 64 + mt * 16 + lm) * LDSS;
            const uint4* pH = (const uint4*)&sAh[mrow + offA];
            const uint4* pL = (const uint4*)&sAl[mrow + offA];
            ah[mt].q[0] = pH[0]; ah[mt].q[1] = pH[1];   // +8 halves = +1 uint4
            al[mt].q[0] = pL[0]; al[mt].q[1] = pL[1];
        }
#pragma unroll
        for (int nt = 0; nt < 2; ++nt) {
            int nrow = (wn * 32 + nt * 16 + lm) * LDSS;
            const uint4* pH = (const uint4*)&sBh[nrow + offB];
            const uint4* pL = (const uint4*)&sBl[nrow + offB];
            bh[nt].q[0] = pH[0]; bh[nt].q[1] = pH[1];
            bl[nt].q[0] = pL[0]; bl[nt].q[1] = pL[1];
        }
#pragma unroll
        for (int mt = 0; mt < 4; ++mt) {
#pragma unroll
            for (int nt = 0; nt < 2; ++nt) {
                acc[mt][nt] = __builtin_amdgcn_wmma_f32_16x16x32_f16(
                    false, ah[mt].v, false, bh[nt].v, (short)0, acc[mt][nt], false, false);
                acc[mt][nt] = __builtin_amdgcn_wmma_f32_16x16x32_f16(
                    false, ah[mt].v, false, bl[nt].v, (short)0, acc[mt][nt], false, false);
                acc[mt][nt] = __builtin_amdgcn_wmma_f32_16x16x32_f16(
                    false, al[mt].v, false, bh[nt].v, (short)0, acc[mt][nt], false, false);
            }
        }

        // ---- stage next tile into the opposite buffer ----
        if (has_next) {
            _Float16* dAh = smem + (cur ^ 1) * BUFH;
            stage_tiles(dAh, dAh + TILEH, dAh + 2 * TILEH, dAh + 3 * TILEH, ra, rb, t);
        }
        __syncthreads();   // single barrier per iteration
    }

    // ---- epilogue: C/D layout (VGPR r -> row r + hiHalf*8, col = lane%16) ----
#pragma unroll
    for (int nt = 0; nt < 2; ++nt) {
        int col = nBase + wn * 32 + nt * 16 + lm;
        float b = bias[col];
#pragma unroll
        for (int mt = 0; mt < 4; ++mt) {
            int rowBase = mBase + wm * 64 + mt * 16 + hiHalf * 8;
#pragma unroll
            for (int r = 0; r < 8; ++r)
                out[(size_t)(rowBase + r) * OUT_F + col] = acc[mt][nt][r] + b;
        }
    }
}

// ---------------------------------------------------------------------------
extern "C" void kernel_launch(void* const* d_in, const int* in_sizes, int n_in,
                              void* d_out, int out_size, void* d_ws, size_t ws_size,
                              hipStream_t stream) {
    const float* x      = (const float*)d_in[0];
    const float* weight = (const float*)d_in[1];
    const float* bias   = (const float*)d_in[2];
    const float* sw     = (const float*)d_in[3];
    const int*   sidx   = (const int*)d_in[4];
    float* out = (float*)d_out;

    float* wsw = (float*)d_ws;                 // fp32 W' copy, 64 MB (L2-resident)
    const int nnz = in_sizes[3];               // 262144
    const int M   = in_sizes[0] / IN_F;        // 8192

    // allow 80KB dynamic LDS (default cap is 64KB; CDNA5 WGP has 320KB)
    hipFuncSetAttribute((const void*)gemm_kernel,
                        hipFuncAttributeMaxDynamicSharedMemorySize, LDS_BYTES);

    wcopy_kernel<<<(OUT_F * IN_F / 4 + 255) / 256, 256, 0, stream>>>(wsw, weight);
    scatter_kernel<<<(nnz + 255) / 256, 256, 0, stream>>>(wsw, sw, sidx, nnz);
    dim3 grid(OUT_F / BN, M / BM);
    gemm_kernel<<<grid, 256, LDS_BYTES, stream>>>(x, wsw, bias, out);
}
